// SPCNNClassifier_70222715289815
// MI455X (gfx1250) — matrix operand order repaired and checked by reading, not compiled, not run on hardware
//
#include <hip/hip_runtime.h>
#include <math.h>

// ---------------- problem constants ----------------
#define B_  16
#define T_  1024
#define E_  1024
#define C_  1024
#define H_  512
#define L_  6
#define G_  2048   // 4*H
#define TP_ 1028   // T + 4 (k=5 conv halo)

typedef __attribute__((ext_vector_type(16))) __bf16 v16bf;
typedef __attribute__((ext_vector_type(8)))  float  v8f;

union Frag { float4 q[2]; v16bf v; };

static __device__ __forceinline__ unsigned short f2bf(float f) {
  union { float f; unsigned u; } v; v.f = f;
  unsigned r = v.u + 0x7fffu + ((v.u >> 16) & 1u);   // round-to-nearest-even
  return (unsigned short)(r >> 16);
}
static __device__ __forceinline__ float bf2f(unsigned short h) {
  union { unsigned u; float f; } v; v.u = ((unsigned)h) << 16;
  return v.f;
}
static __device__ __forceinline__ float sigm(float x) { return 1.f / (1.f + expf(-x)); }

static __device__ __forceinline__ v8f wmma_bf16(const Frag& a, const Frag& b, v8f c) {
  // D = A(16x32 bf16) * B(32x16 bf16) + C(16x16 f32)
  return __builtin_amdgcn_wmma_f32_16x16x32_bf16(false, a.v, false, b.v, (short)0, c,
                                                 false, false);
}

// generation-counter grid barrier (counter zeroed by hipMemsetAsync each launch)
static __device__ __forceinline__ void grid_barrier(unsigned* cnt, unsigned nblk, unsigned bar) {
  __syncthreads();
  if (threadIdx.x == 0) {
    __threadfence();
    atomicAdd(cnt, 1u);
    const unsigned tgt = nblk * bar;
    while (atomicAdd(cnt, 0u) < tgt) { __builtin_amdgcn_s_sleep(1); }
    __threadfence();
  }
  __syncthreads();
}

// ---------------- prep kernels ----------------

// embeddings [B,T,E] f32 -> embp [B, T+4, E] bf16 (2-step zero halo each side)
__global__ void k_pad_embed(const float* __restrict__ src, unsigned short* __restrict__ dst) {
  size_t id = (size_t)blockIdx.x * 256 + threadIdx.x;
  if (id >= (size_t)B_ * TP_ * E_) return;
  int e = (int)(id % E_);
  size_t r = id / E_;
  int tt = (int)(r % TP_);
  int b  = (int)(r / TP_);
  float v = 0.f;
  if (tt >= 2 && tt < T_ + 2) v = src[((size_t)b * T_ + (tt - 2)) * E_ + e];
  dst[id] = f2bf(v);
}

// generic B-matrix pack: src[n*K+k] f32 -> WMMA bf16 B-tile layout
// B 32x16 layout: lane = ((k_in_tile>>3)&1)*16+n_in_tile, slot i = (k_in_tile&7)|((k_in_tile>>4)<<3)
__global__ void k_pack_b(const float* __restrict__ src, unsigned short* __restrict__ dst,
                         int K, int N) {
  size_t id = (size_t)blockIdx.x * 256 + threadIdx.x;
  if (id >= (size_t)K * N) return;
  int k = (int)(id % K), n = (int)(id / K);
  int kt = k >> 5, kk = k & 31, nt = n >> 4, nn = n & 15;
  int lane = (((kk >> 3) & 1) << 4) | nn, i = (kk & 7) | ((kk >> 4) << 3);
  int ntiles = N >> 4;
  dst[(((size_t)kt * ntiles + nt) * 32 + lane) * 16 + i] = f2bf(src[(size_t)n * K + k]);
}

// conv_w [C,E,5] -> packed B with K folded as (tap*32 + kt), N = C
__global__ void k_pack_conv(const float* __restrict__ w, unsigned short* __restrict__ dst) {
  size_t id = (size_t)blockIdx.x * 256 + threadIdx.x;
  if (id >= (size_t)5 * E_ * C_) return;
  int tap = (int)(id / ((size_t)E_ * C_));
  size_t r = id % ((size_t)E_ * C_);
  int k = (int)(r % E_), n = (int)(r / E_);
  int kt = k >> 5, kk = k & 31, nt = n >> 4, nn = n & 15;
  int lane = (((kk >> 3) & 1) << 4) | nn, i = (kk & 7) | ((kk >> 4) << 3);
  int kte = tap * 32 + kt;
  dst[(((size_t)kte * 64 + nt) * 32 + lane) * 16 + i] =
      f2bf(w[((size_t)n * E_ + k) * 5 + tap]);
}

__global__ void k_bias(const float* __restrict__ bih, const float* __restrict__ bhh,
                       float* __restrict__ out) {
  int id = blockIdx.x * 256 + threadIdx.x;
  if (id < 6 * G_) out[id] = bih[id] + bhh[id];
}

// ---------------- conv as GEMM (+BN+ReLU), software-pipelined ----------------
// grid (Mtiles=1024, 4), block 128 (4 waves). wave -> 4 N-tiles. output xbuf[t][b][c] bf16
// K is flattened as idx = tap*32 + kt (160 steps). Final over-read fragment (idx=160)
// stays inside the workspace (embp is followed by xbuf, convP by wihP) and is unused.
__global__ void k_conv_gemm(const unsigned short* __restrict__ embp,
                            const unsigned short* __restrict__ convP,
                            const float* __restrict__ convb,
                            const float* __restrict__ gam, const float* __restrict__ bet,
                            const float* __restrict__ mea, const float* __restrict__ var,
                            unsigned short* __restrict__ xbuf) {
  const int lane = threadIdx.x & 31, wave = threadIdx.x >> 5;
  const int nn = lane & 15, half = lane >> 4;
  const int mtile = blockIdx.x;
  const int ntbase = (blockIdx.y * 4 + wave) * 4;
  const int m = mtile * 16 + nn;               // A row for this lane
  const int b = m >> 10, t = m & 1023;
  const unsigned short* base = embp + ((size_t)b * TP_ + t) * E_ + half * 8;

  auto loadA = [&](Frag& A, int idx) {
    const unsigned short* ar = base + (size_t)(idx >> 5) * E_ + (idx & 31) * 32;
    A.q[0] = *(const float4*)(ar);
    A.q[1] = *(const float4*)(ar + 16);
  };
  auto loadB = [&](Frag* Bf, int idx) {
#pragma unroll
    for (int j = 0; j < 4; ++j) {
      const float4* bp =
          (const float4*)(convP + (((size_t)idx * 64 + ntbase + j) * 32 + lane) * 16);
      Bf[j].q[0] = bp[0]; Bf[j].q[1] = bp[1];
    }
  };

  v8f acc[4] = {};
  Frag a0, a1, b0[4], b1[4];
  loadA(a0, 0); loadB(b0, 0);
  for (int idx = 0; idx < 160; idx += 2) {
    loadA(a1, idx + 1); loadB(b1, idx + 1);          // prefetch step idx+1
#pragma unroll
    for (int j = 0; j < 4; ++j) acc[j] = wmma_bf16(a0, b0[j], acc[j]);
    loadA(a0, idx + 2); loadB(b0, idx + 2);          // prefetch step idx+2 (benign over-read at 160)
#pragma unroll
    for (int j = 0; j < 4; ++j) acc[j] = wmma_bf16(a1, b1[j], acc[j]);
  }

#pragma unroll
  for (int j = 0; j < 4; ++j) {
    const int n = (ntbase + j) * 16 + nn;
    const float s  = gam[n] * rsqrtf(var[n] + 1e-5f);
    const float sh = bet[n] - mea[n] * s;
#pragma unroll
    for (int r = 0; r < 8; ++r) {
      const int m2 = mtile * 16 + half * 8 + r;   // C/D layout: M = half*8 + vgpr
      const int b2 = m2 >> 10, t2 = m2 & 1023;
      float v = acc[j][r] + convb[n];
      v = fmaxf(v * s + sh, 0.f);
      xbuf[((size_t)t2 * B_ + b2) * 1024 + n] = f2bf(v);
    }
  }
}

// ---------------- gates = x @ Wih^T + bias (all T at once), pipelined ----------------
// grid (1024, 8), block 128. M=16384 rows (t*B+b), N=2048, K=1024, f32 out.
__global__ void k_xw_gemm(const unsigned short* __restrict__ xin,
                          const unsigned short* __restrict__ wihP,
                          const float* __restrict__ bias,
                          float* __restrict__ gates) {
  const int lane = threadIdx.x & 31, wave = threadIdx.x >> 5;
  const int nn = lane & 15, half = lane >> 4;
  const int mtile = blockIdx.x;
  const int ntbase = (blockIdx.y * 4 + wave) * 4;
  const unsigned short* arow = xin + (size_t)(mtile * 16 + nn) * 1024 + half * 8;

  auto loadA = [&](Frag& A, int kt) {
    A.q[0] = *(const float4*)(arow + kt * 32);
    A.q[1] = *(const float4*)(arow + kt * 32 + 16);
  };
  auto loadB = [&](Frag* Bf, int kt) {
#pragma unroll
    for (int j = 0; j < 4; ++j) {
      const float4* bp =
          (const float4*)(wihP + (((size_t)kt * 128 + ntbase + j) * 32 + lane) * 16);
      Bf[j].q[0] = bp[0]; Bf[j].q[1] = bp[1];
    }
  };

  v8f acc[4] = {};
  Frag a0, a1, b0[4], b1[4];
  loadA(a0, 0); loadB(b0, 0);
  for (int kt = 0; kt < 32; kt += 2) {
    loadA(a1, kt + 1); loadB(b1, kt + 1);
#pragma unroll
    for (int j = 0; j < 4; ++j) acc[j] = wmma_bf16(a0, b0[j], acc[j]);
    loadA(a0, kt + 2); loadB(b0, kt + 2);            // benign over-read at kt==32
#pragma unroll
    for (int j = 0; j < 4; ++j) acc[j] = wmma_bf16(a1, b1[j], acc[j]);
  }

#pragma unroll
  for (int j = 0; j < 4; ++j) {
    const int n = (ntbase + j) * 16 + nn;
    const float bs = bias[n];
#pragma unroll
    for (int r = 0; r < 8; ++r) {
      const int m2 = mtile * 16 + half * 8 + r;
      gates[(size_t)m2 * G_ + n] = acc[j][r] + bs;
    }
  }
}

// ---------------- sequential LSTM scan ----------------
// 8 WGs x 256 thr. WG wg owns hidden units [wg*64, wg*64+64). Whh slice (256KB) in LDS.
__global__ void k_lstm_scan(const float* __restrict__ gates,
                            const unsigned short* __restrict__ whhP,
                            unsigned short* __restrict__ hbuf,   // [2][16][512] bf16
                            unsigned short* __restrict__ xout,   // +dir*512 col base
                            unsigned* __restrict__ counter,
                            int reverse) {
  extern __shared__ char smem[];
  unsigned short* whh_s = (unsigned short*)smem;            // 16 tiles x 16kt x 32 x 16 = 256KB
  float* g_s = (float*)(smem + 262144);                     // [4][16][64]  16KB
  float* c_s = (float*)(smem + 262144 + 16384);             // [16][64]      4KB

  const int tid = threadIdx.x, lane = tid & 31, wave = tid >> 5;
  const int wg = blockIdx.x, j0 = wg * 64;
  const int nn = lane & 15, half = lane >> 4;
  const int li0 = wave * 2, li1 = wave * 2 + 1;

  // preload this WG's Whh B-tiles into LDS: local tile li -> nt = g4*32 + wg*4 + sub
  for (int li = li0; li <= li1; ++li) {
    const int g4 = li >> 2, sub = li & 3;
    const int nt = g4 * 32 + wg * 4 + sub;
    for (int kt = 0; kt < 16; ++kt) {
      const float4* src = (const float4*)(whhP + (((size_t)kt * 128 + nt) * 32 + lane) * 16);
      float4* dst = (float4*)(whh_s + (((size_t)li * 16 + kt) * 32 + lane) * 16);
      dst[0] = src[0]; dst[1] = src[1];
    }
  }
  for (int i = tid; i < 1024; i += 256) c_s[i] = 0.f;
  for (int i = tid; i < 1024; i += 256) hbuf[wg * 1024 + i] = 0;  // zero h(-1), buffer 0
  __threadfence();
  unsigned bar = 0;
  grid_barrier(counter, gridDim.x, ++bar);

  for (int step = 0; step < T_; ++step) {
    const int tt = reverse ? (T_ - 1 - step) : step;
    const unsigned short* hprev = hbuf + (size_t)(step & 1) * (B_ * H_);
    const unsigned short* arow = hprev + (size_t)nn * H_ + half * 8;

    auto loadA = [&](Frag& A, int kt) {
      A.q[0] = *(const float4*)(arow + kt * 32);
      A.q[1] = *(const float4*)(arow + kt * 32 + 16);
    };
    auto loadB = [&](Frag& Bf, int li, int kt) {
      const float4* p = (const float4*)(whh_s + (((size_t)li * 16 + kt) * 32 + lane) * 16);
      Bf.q[0] = p[0]; Bf.q[1] = p[1];
    };

    v8f acc0 = {}, acc1 = {};
    Frag a0, a1, b00, b01, b10, b11;
    loadA(a0, 0); loadB(b00, li0, 0); loadB(b01, li1, 0);
    for (int kt = 0; kt < 16; kt += 2) {
      loadA(a1, kt + 1); loadB(b10, li0, kt + 1); loadB(b11, li1, kt + 1);
      acc0 = wmma_bf16(a0, b00, acc0);
      acc1 = wmma_bf16(a0, b01, acc1);
      loadA(a0, kt + 2); loadB(b00, li0, kt + 2); loadB(b01, li1, kt + 2); // benign over-read at 16
      acc0 = wmma_bf16(a1, b10, acc0);
      acc1 = wmma_bf16(a1, b11, acc1);
    }

    // add precomputed x-gates, scatter to LDS in (gate, batch, j) order
#pragma unroll
    for (int which = 0; which < 2; ++which) {
      const int li = wave * 2 + which;
      const int g4 = li >> 2, sub = li & 3;
      const int jloc = sub * 16 + nn;
      const v8f& acc = which ? acc1 : acc0;
#pragma unroll
      for (int r = 0; r < 8; ++r) {
        const int bb = half * 8 + r;
        const float v = acc[r] +
            gates[((size_t)tt * B_ + bb) * G_ + (size_t)g4 * H_ + j0 + jloc];
        g_s[((size_t)g4 * 16 + bb) * 64 + jloc] = v;
      }
    }
    __syncthreads();
    // elementwise LSTM cell for this WG's 16x64 hidden block
#pragma unroll
    for (int u = 0; u < 4; ++u) {
      const int idx = tid + u * 256;
      const int bb = idx >> 6, j = idx & 63;
      const float gi = g_s[(0 * 16 + bb) * 64 + j];
      const float gf = g_s[(1 * 16 + bb) * 64 + j];
      const float gg = g_s[(2 * 16 + bb) * 64 + j];
      const float go = g_s[(3 * 16 + bb) * 64 + j];
      const float c = sigm(gf) * c_s[bb * 64 + j] + sigm(gi) * tanhf(gg);
      c_s[bb * 64 + j] = c;
      const unsigned short hb = f2bf(sigm(go) * tanhf(c));
      hbuf[(size_t)((step + 1) & 1) * (B_ * H_) + bb * H_ + j0 + j] = hb;
      xout[((size_t)tt * B_ + bb) * 1024 + j0 + j] = hb;
    }
    __threadfence();
    grid_barrier(counter, gridDim.x, ++bar);
  }
}

// ---------------- classifier (L=6, trivial FLOPs) ----------------
__global__ void k_classifier(const unsigned short* __restrict__ x,
                             const float* __restrict__ w, const float* __restrict__ b,
                             float* __restrict__ logits) {
  const int id = blockIdx.x * 256 + threadIdx.x;   // T*B*L = 98304
  if (id >= T_ * B_ * L_) return;
  const int row = id / L_, l = id % L_;
  const unsigned short* xr = x + (size_t)row * 1024;
  const float* wr = w + (size_t)l * 1024;
  float acc = b[l];
  for (int k = 0; k < 1024; ++k) acc += bf2f(xr[k]) * wr[k];
  logits[id] = acc;
}

// ---------------- CRF NLL (one block) ----------------
__global__ void k_crf(const float* __restrict__ logits, const int* __restrict__ am,
                      const int* __restrict__ lab, const float* __restrict__ start,
                      const float* __restrict__ endw, const float* __restrict__ trans,
                      float* __restrict__ out) {
  __shared__ float tr[36], st[6], en[6], alpha[16 * 6], numv[16], denv[16];
  const int tid = threadIdx.x;
  if (tid < 36) tr[tid] = trans[tid];
  if (tid < 6) { st[tid] = start[tid]; en[tid] = endw[tid]; }
  __syncthreads();
  // numerator (per batch)
  if (tid < 16) {
    const int b = tid;
    int l0 = lab[b * T_];
    int t0 = (l0 == -100) ? 0 : l0;
    float score = st[t0] + logits[(0 * B_ + b) * L_ + t0];
    int prev = t0;
    for (int t = 1; t < T_; ++t) {
      const int lt = lab[b * T_ + t];
      const int mt = (am[b * T_ + t] != 0) && (lt != -100);
      const int tg = (lt == -100) ? 0 : lt;
      if (mt) {
        score += tr[prev * 6 + tg] + logits[((size_t)t * B_ + b) * L_ + tg];
        prev = tg;
      }
    }
    numv[b] = score + en[prev];
  }
  __syncthreads();
  // denominator: forward algorithm, threads (b,j) for idx<96
  const int b = tid / 6, j = tid % 6;
  if (tid < 96) alpha[b * 6 + j] = st[j] + logits[(0 * B_ + b) * L_ + j];
  __syncthreads();
  for (int t = 1; t < T_; ++t) {
    float na = 0.f;
    if (tid < 96) {
      const int lt = lab[b * T_ + t];
      const int mt = (am[b * T_ + t] != 0) && (lt != -100);
      if (mt) {
        float mx = -3.4e38f;
        for (int i = 0; i < 6; ++i) mx = fmaxf(mx, alpha[b * 6 + i] + tr[i * 6 + j]);
        float s = 0.f;
        for (int i = 0; i < 6; ++i) s += expf(alpha[b * 6 + i] + tr[i * 6 + j] - mx);
        na = mx + logf(s) + logits[((size_t)t * B_ + b) * L_ + j];
      } else {
        na = alpha[b * 6 + j];
      }
    }
    __syncthreads();
    if (tid < 96) alpha[b * 6 + j] = na;
    __syncthreads();
  }
  if (tid < 16) {
    float mx = -3.4e38f;
    for (int i = 0; i < 6; ++i) mx = fmaxf(mx, alpha[tid * 6 + i] + en[i]);
    float s = 0.f;
    for (int i = 0; i < 6; ++i) s += expf(alpha[tid * 6 + i] + en[i] - mx);
    denv[tid] = mx + logf(s);
  }
  __syncthreads();
  if (tid == 0) {
    float s = 0.f;
    for (int i = 0; i < 16; ++i) s += numv[i] - denv[i];
    out[0] = -(s / 16.0f);
  }
}

// ---------------- workspace layout ----------------
static constexpr size_t SZ_EMBP  = (size_t)B_ * TP_ * E_ * 2;     // 33,685,504
static constexpr size_t SZ_X     = (size_t)T_ * B_ * 1024 * 2;    // 33,554,432
static constexpr size_t SZ_GATES = (size_t)T_ * B_ * G_ * 4;      // 134,217,728
static constexpr size_t SZ_CONVP = (size_t)5 * E_ * C_ * 2;
static constexpr size_t SZ_WIHP  = (size_t)6 * 1024 * G_ * 2;
static constexpr size_t SZ_WHHP  = (size_t)6 * H_ * G_ * 2;
static constexpr size_t SZ_BIAS  = (size_t)6 * G_ * 4;
static constexpr size_t SZ_H     = (size_t)2 * B_ * H_ * 2;
static constexpr size_t SZ_LOG   = (size_t)T_ * B_ * L_ * 4;
static constexpr size_t SZ_MISC  = 4096;

static constexpr size_t OFF_EMBP  = 0;
static constexpr size_t OFF_XBUF  = OFF_EMBP  + SZ_EMBP;
static constexpr size_t OFF_XNEXT = OFF_XBUF  + SZ_X;
static constexpr size_t OFF_GATES = OFF_XNEXT + SZ_X;
static constexpr size_t OFF_CONVP = OFF_GATES + SZ_GATES;
static constexpr size_t OFF_WIHP  = OFF_CONVP + SZ_CONVP;
static constexpr size_t OFF_WHHP  = OFF_WIHP  + SZ_WIHP;
static constexpr size_t OFF_BIAS  = OFF_WHHP  + SZ_WHHP;
static constexpr size_t OFF_H     = OFF_BIAS  + SZ_BIAS;
static constexpr size_t OFF_LOG   = OFF_H     + SZ_H;
static constexpr size_t OFF_MISC  = OFF_LOG   + SZ_LOG;

extern "C" void kernel_launch(void* const* d_in, const int* in_sizes, int n_in,
                              void* d_out, int out_size, void* d_ws, size_t ws_size,
                              hipStream_t stream) {
  (void)in_sizes; (void)n_in; (void)out_size; (void)ws_size;
  char* ws = (char*)d_ws;
  const float* emb    = (const float*)d_in[0];
  const int*   am     = (const int*)d_in[1];
  const int*   lab    = (const int*)d_in[2];
  const float* conv_w = (const float*)d_in[3];
  const float* conv_b = (const float*)d_in[4];
  const float* bn_g   = (const float*)d_in[5];
  const float* bn_b   = (const float*)d_in[6];
  const float* bn_m   = (const float*)d_in[7];
  const float* bn_v   = (const float*)d_in[8];
  const float* Wih    = (const float*)d_in[9];
  const float* Whh    = (const float*)d_in[10];
  const float* bih    = (const float*)d_in[11];
  const float* bhh    = (const float*)d_in[12];
  const float* cls_w  = (const float*)d_in[13];
  const float* cls_b  = (const float*)d_in[14];
  const float* crf_s  = (const float*)d_in[15];
  const float* crf_e  = (const float*)d_in[16];
  const float* crf_t  = (const float*)d_in[17];

  unsigned short* embp  = (unsigned short*)(ws + OFF_EMBP);
  unsigned short* xbuf  = (unsigned short*)(ws + OFF_XBUF);
  unsigned short* xnext = (unsigned short*)(ws + OFF_XNEXT);
  float*          gates = (float*)(ws + OFF_GATES);
  unsigned short* convP = (unsigned short*)(ws + OFF_CONVP);
  unsigned short* wihP  = (unsigned short*)(ws + OFF_WIHP);
  unsigned short* whhP  = (unsigned short*)(ws + OFF_WHHP);
  float*          bias  = (float*)(ws + OFF_BIAS);
  unsigned short* hbuf  = (unsigned short*)(ws + OFF_H);
  float*          logit = (float*)(ws + OFF_LOG);

  hipMemsetAsync(ws + OFF_MISC, 0, SZ_MISC, stream);

  {
    size_t n = (size_t)B_ * TP_ * E_;
    k_pad_embed<<<(unsigned)((n + 255) / 256), 256, 0, stream>>>(emb, embp);
  }
  {
    size_t n = (size_t)5 * E_ * C_;
    k_pack_conv<<<(unsigned)((n + 255) / 256), 256, 0, stream>>>(conv_w, convP);
  }
  for (int ld = 0; ld < 6; ++ld) {
    k_pack_b<<<8192, 256, 0, stream>>>(Wih + (size_t)ld * G_ * 1024,
                                       wihP + (size_t)ld * G_ * 1024, 1024, G_);
    k_pack_b<<<4096, 256, 0, stream>>>(Whh + (size_t)ld * G_ * H_,
                                       whhP + (size_t)ld * G_ * H_, H_, G_);
  }
  k_bias<<<48, 256, 0, stream>>>(bih, bhh, bias);

  k_conv_gemm<<<dim3(1024, 4), 128, 0, stream>>>(embp, convP, conv_b, bn_g, bn_b,
                                                 bn_m, bn_v, xbuf);

  const size_t scanLds = 262144 + 16384 + 4096;   // Whh slice + gates + cell state
  for (int l = 0; l < 3; ++l) {
    const unsigned short* xin = (l & 1) ? xnext : xbuf;   // l0:xbuf  l1:xnext  l2:xbuf
    unsigned short* xo        = (l & 1) ? xbuf  : xnext;
    for (int d = 0; d < 2; ++d) {
      const int ld = l * 2 + d;
      k_xw_gemm<<<dim3(1024, 8), 128, 0, stream>>>(
          xin, wihP + (size_t)ld * G_ * 1024, bias + (size_t)ld * G_, gates);
      unsigned* cnt = (unsigned*)(ws + OFF_MISC + (size_t)ld * 256);
      k_lstm_scan<<<8, 256, scanLds, stream>>>(
          gates, whhP + (size_t)ld * G_ * H_, hbuf, xo + (size_t)d * H_, cnt, d);
    }
  }

  k_classifier<<<384, 256, 0, stream>>>(xnext, cls_w, cls_b, logit);
  k_crf<<<1, 128, 0, stream>>>(logit, am, lab, crf_s, crf_e, crf_t, (float*)d_out);
}
